// TemporalHiLoWithPositionalEncoding_49306224558681
// MI455X (gfx1250) — matrix operand order repaired and checked by reading, not compiled, not run on hardware
//
#include <hip/hip_runtime.h>

typedef __attribute__((ext_vector_type(16))) __bf16 v16bf;
typedef __attribute__((ext_vector_type(8)))  float  v8f;
typedef unsigned short ushort_t;
typedef unsigned int   uint32;

#define DIMC      512
#define HEAD_DIM  64
#define SEQ       1024
#define BATCH     16
#define MTOK      (BATCH * SEQ)     // 16384
#define ATT_SCALE 0.125f            // 64^-0.5

__device__ __forceinline__ ushort_t f2bf(float f) {
  uint32 u = __float_as_uint(f);
  u += 0x7FFFu + ((u >> 16) & 1u);   // round-to-nearest-even
  return (ushort_t)(u >> 16);
}
__device__ __forceinline__ float bf2f(ushort_t h) {
  return __uint_as_float(((uint32)h) << 16);
}

// Rotate-based all-lanes reduction across each 16-lane DPP row (pure VALU).
__device__ __forceinline__ float row16_max(float x) {
  int t;
  t = __builtin_amdgcn_update_dpp(0, __float_as_int(x), 0x121, 0xF, 0xF, true);
  x = fmaxf(x, __int_as_float(t));   // row_ror:1
  t = __builtin_amdgcn_update_dpp(0, __float_as_int(x), 0x122, 0xF, 0xF, true);
  x = fmaxf(x, __int_as_float(t));   // row_ror:2
  t = __builtin_amdgcn_update_dpp(0, __float_as_int(x), 0x124, 0xF, 0xF, true);
  x = fmaxf(x, __int_as_float(t));   // row_ror:4
  t = __builtin_amdgcn_update_dpp(0, __float_as_int(x), 0x128, 0xF, 0xF, true);
  x = fmaxf(x, __int_as_float(t));   // row_ror:8
  return x;
}
__device__ __forceinline__ float row16_sum(float x) {
  int t;
  t = __builtin_amdgcn_update_dpp(0, __float_as_int(x), 0x121, 0xF, 0xF, true);
  x += __int_as_float(t);
  t = __builtin_amdgcn_update_dpp(0, __float_as_int(x), 0x122, 0xF, 0xF, true);
  x += __int_as_float(t);
  t = __builtin_amdgcn_update_dpp(0, __float_as_int(x), 0x124, 0xF, 0xF, true);
  x += __int_as_float(t);
  t = __builtin_amdgcn_update_dpp(0, __float_as_int(x), 0x128, 0xF, 0xF, true);
  x += __int_as_float(t);
  return x;
}

union Frag {
  v16bf v;
  uint4 q[2];
  ushort_t u[16];
};

// ---------------------------------------------------------------- preprocess
__global__ __launch_bounds__(256, 1) void k_addpos_cast(
    const float* __restrict__ x, const float* __restrict__ pos,
    ushort_t* __restrict__ xbf) {
  int i  = blockIdx.x * blockDim.x + threadIdx.x;       // over MTOK*DIMC
  int nc = i % (SEQ * DIMC);                            // pos is (1,1024,512)
  xbf[i] = f2bf(x[i] + pos[nc]);
}

__global__ __launch_bounds__(256, 1) void k_cast(
    const float* __restrict__ in, ushort_t* __restrict__ out, int n) {
  int i = blockIdx.x * blockDim.x + threadIdx.x;
  if (i < n) out[i] = f2bf(in[i]);
}

// ---------------------------------------------------------------- WMMA GEMM
// C[M,N] = A[M,K](bf16) @ B[K,N](bf16) (+bias). Block tile 64x64, BK=32.
// 128 threads = 4 waves in 2x2; each wave owns a 32x32 patch (2x2 WMMA accs).
template <bool OUT_BF16>
__global__ __launch_bounds__(128, 1) void k_gemm(
    const ushort_t* __restrict__ A, int lda,
    const ushort_t* __restrict__ B, int ldb,
    const float* __restrict__ bias,
    void* __restrict__ Cv, int ldc, int K) {
  __shared__ __align__(16) ushort_t As[64][32];   // [m][k]
  __shared__ __align__(16) ushort_t Bs[64][32];   // [n][k]  (transposed)

  const int tid  = threadIdx.x;
  const int lane = tid & 31;
  const int wave = tid >> 5;
  const int m0 = blockIdx.y * 64;
  const int n0 = blockIdx.x * 64;
  const int wm = (wave >> 1) * 32;
  const int wn = (wave & 1) * 32;

  const v8f zacc = {0.f, 0.f, 0.f, 0.f, 0.f, 0.f, 0.f, 0.f};
  v8f acc[2][2];
  for (int i = 0; i < 2; i++)
    for (int j = 0; j < 2; j++) acc[i][j] = zacc;

  // hoisted, strength-reduced tile pointers
  const int arow = tid >> 1, aoff = (tid & 1) * 16;
  const int bk   = tid >> 2, boff = (tid & 3) * 16;
  const ushort_t* aptr = A + (size_t)(m0 + arow) * lda + aoff;
  const ushort_t* bptr = B + (size_t)bk * ldb + n0 + boff;
  const size_t bstep = 32 * (size_t)ldb;

  for (int kk = 0; kk < K; kk += 32, aptr += 32, bptr += bstep) {
    __syncthreads();
    {   // A tile: 64 rows x 32 k, 16 halfs per thread
      *(uint4*)&As[arow][aoff]     = *(const uint4*)aptr;
      *(uint4*)&As[arow][aoff + 8] = *(const uint4*)(aptr + 8);
    }
    {   // B tile: coalesced read of B[k][n], scatter transposed into Bs[n][k]
      uint4 p0 = *(const uint4*)bptr;
      uint4 p1 = *(const uint4*)(bptr + 8);
      const ushort_t* h0 = (const ushort_t*)&p0;
      const ushort_t* h1 = (const ushort_t*)&p1;
#pragma unroll
      for (int j = 0; j < 8; j++) {
        Bs[boff + j][bk]     = h0[j];
        Bs[boff + 8 + j][bk] = h1[j];
      }
    }
    __syncthreads();

    Frag a[2];
#pragma unroll
    for (int mi = 0; mi < 2; mi++) {
      // 16-bit A layout: lanes 0-15 K={0..7,16..23}; lanes 16-31 K+8
      const ushort_t* p = &As[wm + mi * 16 + (lane & 15)][(lane >> 4) * 8];
      a[mi].q[0] = *(const uint4*)p;
      a[mi].q[1] = *(const uint4*)(p + 16);
    }
#pragma unroll
    for (int ni = 0; ni < 2; ni++) {
      // B layout: lanes 0-15 K={0..7,16..23}; lanes 16-31 K+8 (n-major LDS)
      Frag b;
      const ushort_t* p = &Bs[wn + ni * 16 + (lane & 15)][(lane >> 4) * 8];
      b.q[0] = *(const uint4*)p;
      b.q[1] = *(const uint4*)(p + 16);
#pragma unroll
      for (int mi = 0; mi < 2; mi++)
        acc[mi][ni] = __builtin_amdgcn_wmma_f32_16x16x32_bf16(
            false, a[mi].v, false, b.v, (short)0, acc[mi][ni], false, false);
    }
  }

#pragma unroll
  for (int mi = 0; mi < 2; mi++)
#pragma unroll
    for (int ni = 0; ni < 2; ni++) {
      int row0 = m0 + wm + mi * 16 + ((lane >> 4) ? 8 : 0);
      int col  = n0 + wn + ni * 16 + (lane & 15);
      float bv = bias ? bias[col] : 0.f;
#pragma unroll
      for (int r = 0; r < 8; r++) {
        float v = acc[mi][ni][r] + bv;
        size_t idx = (size_t)(row0 + r) * ldc + col;
        if (OUT_BF16) ((ushort_t*)Cv)[idx] = f2bf(v);
        else          ((float*)Cv)[idx]    = v;
      }
    }
}

// ------------------------------------------------------ flash attention (WMMA)
// 4 waves per block, all on the same (b, head); wave w owns q-tile (qt4*4+w).
// Shared V-tile staging (transposed in LDS) is amortized over the 4 waves.
__global__ __launch_bounds__(128, 1) void k_flash(
    const ushort_t* __restrict__ Qp, int ldq,
    const ushort_t* __restrict__ Kp, int ldk,
    const ushort_t* __restrict__ Vp, int ldv,
    ushort_t* __restrict__ Op, int ldo, int n_heads) {
  __shared__ __align__(16) ushort_t Pt[4][16][32];  // per-wave P staging
  __shared__ __align__(16) ushort_t Vt[64][32];     // shared V^T [d][key]

  const int tid  = threadIdx.x;
  const int lane = tid & 31;
  const int wave = tid >> 5;
  int bx  = blockIdx.x;
  int qt4 = bx & 15;                                // 16 blocks cover 64 q-tiles
  int h   = (bx >> 4) % n_heads;
  int b   = bx / (16 * n_heads);

  const int tok0 = b * SEQ + (qt4 * 4 + wave) * 16;
  const int hc   = h * HEAD_DIM;

  Frag qa[2];   // Q rows, d in [0,32) and [32,64)
  {
    const ushort_t* qrow =
        Qp + (size_t)(tok0 + (lane & 15)) * ldq + hc + ((lane >> 4) ? 8 : 0);
#pragma unroll
    for (int t = 0; t < 2; t++) {
      qa[t].q[0] = *(const uint4*)(qrow + t * 32);
      qa[t].q[1] = *(const uint4*)(qrow + t * 32 + 16);
    }
  }

  const v8f zacc = {0.f, 0.f, 0.f, 0.f, 0.f, 0.f, 0.f, 0.f};
  v8f o[4];
#pragma unroll
  for (int t = 0; t < 4; t++) o[t] = zacc;
  float mrow[8], lrow[8];
#pragma unroll
  for (int r = 0; r < 8; r++) { mrow[r] = -1e30f; lrow[r] = 0.f; }

  // hoisted, strength-reduced pointers (in-loop offsets become immediates)
  const ushort_t* kptr0 =
      Kp + (size_t)(b * SEQ + (lane & 15)) * ldk + hc + ((lane >> 4) ? 8 : 0);
  const ushort_t* kptr1 = kptr0 + 16 * (size_t)ldk;
  const ushort_t* vptr =
      Vp + (size_t)(b * SEQ + (tid >> 2)) * ldv + hc + (tid & 3) * 16;
  const size_t kstep = 32 * (size_t)ldk;
  const size_t vstep = 32 * (size_t)ldv;
  const int vd0 = (tid & 3) * 16;
  const int vkey = tid >> 2;

  for (int kt = 0; kt < SEQ; kt += 32, kptr0 += kstep, kptr1 += kstep, vptr += vstep) {
    __syncthreads();   // previous iteration's Vt reads complete

    // stage V tile transposed: Vt[d][key], cooperatively by all 128 threads
    {
      uint4 p0 = *(const uint4*)vptr;
      uint4 p1 = *(const uint4*)(vptr + 8);
      const ushort_t* h0 = (const ushort_t*)&p0;
      const ushort_t* h1 = (const ushort_t*)&p1;
#pragma unroll
      for (int j = 0; j < 8; j++) {
        Vt[vd0 + j][vkey]     = h0[j];
        Vt[vd0 + 8 + j][vkey] = h1[j];
      }
    }

    // K fragments for this tile (batched loads -> clauses, overlapped waits)
    Frag kb[2][2];   // [key-half][d-tile]; B: n=key (lane&15), k=d
#pragma unroll
    for (int dt = 0; dt < 2; dt++) {
      kb[0][dt].q[0] = *(const uint4*)(kptr0 + dt * 32);
      kb[0][dt].q[1] = *(const uint4*)(kptr0 + dt * 32 + 16);
      kb[1][dt].q[0] = *(const uint4*)(kptr1 + dt * 32);
      kb[1][dt].q[1] = *(const uint4*)(kptr1 + dt * 32 + 16);
    }

    // S = Q K^T : two 16x16 accs (keys [0,16) / [16,32)), WMMA K-dim = d
    v8f s[2];
    s[0] = zacc; s[1] = zacc;
#pragma unroll
    for (int half = 0; half < 2; half++)
#pragma unroll
      for (int dt = 0; dt < 2; dt++)
        s[half] = __builtin_amdgcn_wmma_f32_16x16x32_bf16(
            false, qa[dt].v, false, kb[half][dt].v, (short)0, s[half], false, false);

    // online softmax (C layout: row = r + 8*(lane>=16), col = lane&15)
#pragma unroll
    for (int r = 0; r < 8; r++) {
      float s0 = s[0][r] * ATT_SCALE;
      float s1 = s[1][r] * ATT_SCALE;
      float lm    = row16_max(fmaxf(s0, s1));
      float mnew  = fmaxf(mrow[r], lm);
      float alpha = __expf(mrow[r] - mnew);
      float p0 = __expf(s0 - mnew);
      float p1 = __expf(s1 - mnew);
      float ps = row16_sum(p0 + p1);
      lrow[r] = lrow[r] * alpha + ps;
      mrow[r] = mnew;
#pragma unroll
      for (int t = 0; t < 4; t++) o[t][r] *= alpha;
      int m = r + ((lane >> 4) ? 8 : 0);
      int n = lane & 15;
      Pt[wave][m][n]      = f2bf(p0);
      Pt[wave][m][n + 16] = f2bf(p1);
    }
    __syncthreads();   // Vt + Pt writes visible

    // O += P(16x32) @ V(32x64)
    Frag pa;
    {
      const ushort_t* p = &Pt[wave][lane & 15][(lane >> 4) * 8];
      pa.q[0] = *(const uint4*)p;
      pa.q[1] = *(const uint4*)(p + 16);
    }
#pragma unroll
    for (int t = 0; t < 4; t++) {
      Frag vb;   // B: n=d (lane&15), k=key; contiguous in Vt rows
      const ushort_t* p = &Vt[t * 16 + (lane & 15)][(lane >> 4) * 8];
      vb.q[0] = *(const uint4*)p;
      vb.q[1] = *(const uint4*)(p + 16);
      o[t] = __builtin_amdgcn_wmma_f32_16x16x32_bf16(
          false, pa.v, false, vb.v, (short)0, o[t], false, false);
    }
  }

#pragma unroll
  for (int t = 0; t < 4; t++)
#pragma unroll
    for (int r = 0; r < 8; r++) {
      int row = tok0 + r + ((lane >> 4) ? 8 : 0);
      int col = hc + t * 16 + (lane & 15);
      Op[(size_t)row * ldo + col] = f2bf(o[t][r] / lrow[r]);
    }
}

// ------------------------------------------- hifi windowed attention (T = 4)
// One wave per (b, window, head). Lane owns dims d and d+32.
__global__ __launch_bounds__(32, 1) void k_hifi(
    const ushort_t* __restrict__ hqkv, ushort_t* __restrict__ out) {
  const int lane = threadIdx.x;
  int bx = blockIdx.x;                    // b*1024 + g*4 + head
  int hh = bx & 3;
  int g  = (bx >> 2) & 255;
  int b  = bx >> 10;
  int gh = g >> 4, gw = g & 15;

  int rows[4];
#pragma unroll
  for (int t = 0; t < 4; t++) {
    int i = t >> 1, j = t & 1;
    rows[t] = b * SEQ + (2 * gh + i) * 32 + (2 * gw + j);
  }

  float q[4][2], k[4][2], v[4][2];
#pragma unroll
  for (int t = 0; t < 4; t++) {
    const ushort_t* rp = hqkv + (size_t)rows[t] * 768 + hh * 64;
    q[t][0] = bf2f(rp[lane]);        q[t][1] = bf2f(rp[lane + 32]);
    k[t][0] = bf2f(rp[256 + lane]);  k[t][1] = bf2f(rp[256 + lane + 32]);
    v[t][0] = bf2f(rp[512 + lane]);  v[t][1] = bf2f(rp[512 + lane + 32]);
  }

  float sc[4][4];
#pragma unroll
  for (int i = 0; i < 4; i++)
#pragma unroll
    for (int j = 0; j < 4; j++) {
      float p = q[i][0] * k[j][0] + q[i][1] * k[j][1];
      p = row16_sum(p);                       // 16-lane rows (DPP)
      p += __shfl_xor(p, 16, 32);             // cross-row combine
      sc[i][j] = p * ATT_SCALE;
    }

#pragma unroll
  for (int i = 0; i < 4; i++) {
    float m = fmaxf(fmaxf(sc[i][0], sc[i][1]), fmaxf(sc[i][2], sc[i][3]));
    float e0 = __expf(sc[i][0] - m), e1 = __expf(sc[i][1] - m);
    float e2 = __expf(sc[i][2] - m), e3 = __expf(sc[i][3] - m);
    float inv = 1.f / (e0 + e1 + e2 + e3);
    float o0 = (e0 * v[0][0] + e1 * v[1][0] + e2 * v[2][0] + e3 * v[3][0]) * inv;
    float o1 = (e0 * v[0][1] + e1 * v[1][1] + e2 * v[2][1] + e3 * v[3][1]) * inv;
    ushort_t* op = out + (size_t)rows[i] * 256 + hh * 64;
    op[lane]      = f2bf(o0);
    op[lane + 32] = f2bf(o1);
  }
}

// ---------------------------------------------------------------- launcher
extern "C" void kernel_launch(void* const* d_in, const int* in_sizes, int n_in,
                              void* d_out, int out_size, void* d_ws, size_t ws_size,
                              hipStream_t stream) {
  (void)in_sizes; (void)n_in; (void)out_size; (void)ws_size;
  const float* x         = (const float*)d_in[0];
  const float* pos       = (const float*)d_in[1];
  const float* l_q_w     = (const float*)d_in[2];
  const float* l_kv_w    = (const float*)d_in[3];
  const float* l_proj_w  = (const float*)d_in[4];
  const float* l_proj_b  = (const float*)d_in[5];
  const float* h_qkv_w   = (const float*)d_in[6];
  const float* h_proj_w  = (const float*)d_in[7];
  const float* h_proj_b  = (const float*)d_in[8];
  const float* in_proj_w = (const float*)d_in[9];
  const float* in_proj_b = (const float*)d_in[10];
  const float* out_proj_w= (const float*)d_in[11];
  const float* out_proj_b= (const float*)d_in[12];

  ushort_t* ws = (ushort_t*)d_ws;
  // region0 (48MB): x_bf | h_qkv | l_q   -> reused later as mha qkv (16384x1536)
  ushort_t* xbf  = ws;                       // 16384*512
  ushort_t* hqkv = ws + 8388608;             // 16384*768
  ushort_t* lq   = ws + 20971520;            // 16384*256
  ushort_t* qkvm = ws;                       // 16384*1536 (reuse region0)
  // region1 (16MB): l_kv -> reused as mha attention output
  ushort_t* lkv  = ws + 25165824;            // 16384*512
  ushort_t* mhao = ws + 25165824;
  // region2/3
  ushort_t* hfa  = ws + 33554432;            // 16384*256
  ushort_t* lfa  = ws + 37748736;            // 16384*256
  ushort_t* ybf  = ws + 41943040;            // 16384*512
  // bf16 weights
  ushort_t* wb     = ws + 50331648;
  ushort_t* w_lq   = wb;                     // 131072
  ushort_t* w_lkv  = wb + 131072;            // 262144
  ushort_t* w_lp   = wb + 393216;            // 65536
  ushort_t* w_hqkv = wb + 458752;            // 393216
  ushort_t* w_hp   = wb + 851968;            // 65536
  ushort_t* w_in   = wb + 917504;            // 786432
  ushort_t* w_out  = wb + 1703936;           // 262144

  // 1) x + pos -> bf16
  k_addpos_cast<<<(MTOK * DIMC) / 256, 256, 0, stream>>>(x, pos, xbf);
  // 2) weights -> bf16
  k_cast<<<(131072 + 255) / 256, 256, 0, stream>>>(l_q_w,     w_lq,   131072);
  k_cast<<<(262144 + 255) / 256, 256, 0, stream>>>(l_kv_w,    w_lkv,  262144);
  k_cast<<<( 65536 + 255) / 256, 256, 0, stream>>>(l_proj_w,  w_lp,    65536);
  k_cast<<<(393216 + 255) / 256, 256, 0, stream>>>(h_qkv_w,   w_hqkv, 393216);
  k_cast<<<( 65536 + 255) / 256, 256, 0, stream>>>(h_proj_w,  w_hp,    65536);
  k_cast<<<(786432 + 255) / 256, 256, 0, stream>>>(in_proj_w, w_in,   786432);
  k_cast<<<(262144 + 255) / 256, 256, 0, stream>>>(out_proj_w,w_out,  262144);

  // 3) input projections (WMMA GEMMs)
  k_gemm<true><<<dim3(768 / 64, MTOK / 64), 128, 0, stream>>>(xbf, DIMC, w_hqkv, 768, nullptr, hqkv, 768, DIMC);
  k_gemm<true><<<dim3(256 / 64, MTOK / 64), 128, 0, stream>>>(xbf, DIMC, w_lq,   256, nullptr, lq,   256, DIMC);
  k_gemm<true><<<dim3(512 / 64, MTOK / 64), 128, 0, stream>>>(xbf, DIMC, w_lkv,  512, nullptr, lkv,  512, DIMC);

  // 4) attentions
  k_hifi<<<BATCH * 256 * 4, 32, 0, stream>>>(hqkv, hfa);
  k_flash<<<BATCH * 4 * 16, 128, 0, stream>>>(lq, 256, lkv, 512, lkv + 256, 512, lfa, 256, 4);

  // 5) head projections into y = [hifi | lofi]
  k_gemm<true><<<dim3(256 / 64, MTOK / 64), 128, 0, stream>>>(hfa, 256, w_hp, 256, h_proj_b, ybf,       512, 256);
  k_gemm<true><<<dim3(256 / 64, MTOK / 64), 128, 0, stream>>>(lfa, 256, w_lp, 256, l_proj_b, ybf + 256, 512, 256);

  // 6) MHA
  k_gemm<true><<<dim3(1536 / 64, MTOK / 64), 128, 0, stream>>>(ybf, DIMC, w_in, 1536, in_proj_b, qkvm, 1536, DIMC);
  k_flash<<<BATCH * 8 * 16, 128, 0, stream>>>(qkvm, 1536, qkvm + 512, 1536, qkvm + 1024, 1536, mhao, 512, 8);

  // 7) out projection -> fp32 output
  k_gemm<false><<<dim3(512 / 64, MTOK / 64), 128, 0, stream>>>(mhao, DIMC, w_out, 512, out_proj_b, (float*)d_out, 512, DIMC);
}
